// SynthesisInput_25074019074690
// MI455X (gfx1250) — matrix operand order repaired and checked by reading, not compiled, not run on hardware
//
#include <hip/hip_runtime.h>
#include <hip/hip_bf16.h>
#include <math.h>

typedef _Float16 v16h __attribute__((ext_vector_type(16)));
typedef __fp16   v2fp __attribute__((ext_vector_type(2)));
typedef float    v8f  __attribute__((ext_vector_type(8)));
typedef float    v4f  __attribute__((ext_vector_type(4)));

#define B_   16
#define WD   512
#define C_   1024
#define SZ   64
#define NPIX (SZ*SZ)   // 4096
#define MT   128       // pixel tile
#define NT   128       // out-channel tile
#define KB   32        // K step (WMMA K)
#define NK   (C_/KB)   // 32 K-steps

// workspace layout (float indices from base of d_ws)
static constexpr size_t OFF_PAR = 0;                    // 16*4 floats: c,s,tr0,tr1
static constexpr size_t OFF_FX  = 64;
static constexpr size_t OFF_FY  = OFF_FX + (size_t)B_*C_;
static constexpr size_t OFF_PH  = OFF_FY + (size_t)B_*C_;
static constexpr size_t OFF_AM  = OFF_PH + (size_t)B_*C_;
static constexpr size_t OFF_WT  = OFF_AM + (size_t)B_*C_;   // f16 region starts here

// ---------------- Stage 1: affine -> rotation params per batch ----------------
__global__ void k_affine(const float* __restrict__ w, const float* __restrict__ aw,
                         const float* __restrict__ ab, float* __restrict__ par) {
  __shared__ float tt[B_][4];
  const int t = threadIdx.x;        // 64 threads
  const int b = t >> 2, j = t & 3;
  float acc = 0.f;
  for (int i = 0; i < WD; ++i) acc += w[b*WD + i] * aw[j*WD + i];
  tt[b][j] = acc * 0.04419417382415922f /* 1/sqrt(512) */ + ab[j];
  __syncthreads();
  if (t < B_) {
    float t0 = tt[t][0], t1 = tt[t][1], t2 = tt[t][2], t3 = tt[t][3];
    const float inv = rsqrtf(t0*t0 + t1*t1);
    const float c = t0*inv, s = t1*inv;
    t2 *= inv; t3 *= inv;
    par[t*4+0] = c;
    par[t*4+1] = s;
    par[t*4+2] = -(c*t2 - s*t3);   // trans_col[0]
    par[t*4+3] = -(s*t2 + c*t3);   // trans_col[1]
  }
}

// ---------------- Stage 2: per-(batch,channel) fourier params ----------------
__global__ void k_chan(const float* __restrict__ freqs, const float* __restrict__ phases,
                       const float* __restrict__ par, float* __restrict__ ws) {
  const int idx = blockIdx.x * blockDim.x + threadIdx.x;
  if (idx >= B_*C_) return;
  const int b = idx >> 10, c = idx & (C_-1);
  const float f0 = freqs[c*2+0], f1 = freqs[c*2+1];
  const float rc = par[b*4+0], rs = par[b*4+1];
  const float tr0 = par[b*4+2], tr1 = par[b*4+3];
  const float fb0 =  f0*rc + f1*rs;
  const float fb1 = -f0*rs + f1*rc;
  const float ph  = phases[c] + f0*tr0 + f1*tr1;
  const float nrm = sqrtf(fb0*fb0 + fb1*fb1);
  float amp = 1.0f - (nrm - 2.0f) * (1.0f/30.0f);
  amp = fminf(fmaxf(amp, 0.0f), 1.0f);
  ws[OFF_FX + idx] = fb0;
  ws[OFF_FY + idx] = fb1;
  ws[OFF_PH + idx] = ph;
  ws[OFF_AM + idx] = amp;
}

// ---------------- Stage 3: weight -> f16 transposed, scaled by 1/sqrt(C)=1/32 ----------------
__global__ void k_wt(const float* __restrict__ weight, _Float16* __restrict__ wT) {
  __shared__ float tile[32][33];
  const int tx = threadIdx.x & 31, ty = threadIdx.x >> 5;   // 32x8
  const int k0 = blockIdx.y * 32, c0 = blockIdx.x * 32;
  #pragma unroll
  for (int i = 0; i < 32; i += 8)
    tile[ty+i][tx] = weight[(size_t)(k0+ty+i)*C_ + c0 + tx];
  __syncthreads();
  #pragma unroll
  for (int i = 0; i < 32; i += 8)
    wT[(size_t)(c0+ty+i)*C_ + k0 + tx] = (_Float16)(tile[tx][ty+i] * 0.03125f);
}

// ---------------- Stage 4: fused sin-synthesis + WMMA GEMM (double-buffered, async B) ----------------
// A-tile LDS layout is fragment-permuted: k' = (k&7) | ((k&16)>>1) | ((k&8)<<1),
// so each lane's 16 A-halves are one contiguous 32B run in the ISA WMMA order.
__global__ __launch_bounds__(256)
void k_main(const float* __restrict__ ws, const _Float16* __restrict__ wT,
            float* __restrict__ out) {
  __shared__ _Float16 lA[2][MT*KB];   // 2 x 8 KB : 128 pixels x 32 channels (permuted k')
  __shared__ _Float16 lB[2][KB*NT];   // 2 x 8 KB : 32 channels x 128 outputs, [k][n]
  __shared__ float    lP[4*C_];       // 16 KB    : fx | fy | ph | amp for this batch

  const int tid = threadIdx.x;
  const int b   = blockIdx.z;
  const int p0  = blockIdx.y * MT;
  const int n0  = blockIdx.x * NT;

  // stage the per-batch channel params once
  for (int i = tid; i < C_; i += 256) {
    lP[0*C_ + i] = ws[OFF_FX + (size_t)b*C_ + i];
    lP[1*C_ + i] = ws[OFF_FY + (size_t)b*C_ + i];
    lP[2*C_ + i] = ws[OFF_PH + (size_t)b*C_ + i];
    lP[3*C_ + i] = ws[OFF_AM + (size_t)b*C_ + i];
  }

  // this thread's role in A-tile synthesis: row m_l, half of K range
  const int khalf = tid >> 7;            // 0 or 1 -> k offset 0 / 16
  const int m_l   = tid & 127;
  const int p     = p0 + m_l;
  const float u   = ((p & 63) + 0.5f) * (1.0f/64.0f) - 0.5f;   // sx = 0.5
  const float v   = ((p >> 6) + 0.5f) * (1.0f/64.0f) - 0.5f;   // sy = 0.5

  const int lane = tid & 31;
  const int wid  = tid >> 5;
  const int wm   = wid & 3;              // 4 wave-rows  x 32 pixels
  const int wn   = wid >> 2;             // 2 wave-cols  x 64 channels
  const int ksel = (lane >> 4) << 4;     // lanes 0-15 read k'=0..15, lanes 16-31 read k'=16..31

  // B-tile staging role: 32 consecutive bytes (16 halves) per thread
  const int li = tid * 16;               // half index 0..4095
  const int kr = li >> 7;                // row within K step
  const int nc = li & 127;               // col within N tile

  // kick async copy of weight tile kt into lB[buf] (ASYNCcnt-tracked, LDS-direct)
  auto asyncB = [&](int kt, int buf) {
    const _Float16* g = wT + (size_t)(kt*KB + kr)*C_ + n0 + nc;     // 32B aligned
    const unsigned  l = (unsigned)(uintptr_t)(&lB[buf][li]);        // LDS byte offset
    asm volatile("global_load_async_to_lds_b128 %0, %1, off"
                 :: "v"(l), "v"(g) : "memory");
    asm volatile("global_load_async_to_lds_b128 %0, %1, off offset:16"
                 :: "v"(l), "v"(g) : "memory");
    if (kt + 2 < NK)                     // warm GL2 two tiles ahead
      __builtin_prefetch(g + (size_t)2*KB*C_, 0, 1);
  };

  // synthesize 16 sin-features for K-step kt into lA[buf] (permuted layout).
  // arg is in revolutions -> raw v_sin_f32 via __builtin_amdgcn_sinf.
  auto synthA = [&](int kt, int buf) {
    const int k0 = kt * KB;
    unsigned int pk[8];
    #pragma unroll
    for (int jj = 0; jj < 8; ++jj) {
      float sv[2];
      #pragma unroll
      for (int e = 0; e < 2; ++e) {
        const int c = k0 + khalf*16 + jj*2 + e;
        const float a = u*lP[c] + v*lP[C_ + c] + lP[2*C_ + c];
        sv[e] = __builtin_amdgcn_sinf(a) * lP[3*C_ + c];
      }
      const v2fp h2 = __builtin_amdgcn_cvt_pkrtz(sv[0], sv[1]);  // v_cvt_pk_rtz_f16_f32
      union { v2fp h; unsigned int u; } cv; cv.h = h2;
      pk[jj] = cv.u;
    }
    // permuted chunk offsets: khalf=0 -> k'={0..7,16..23}; khalf=1 -> k'={8..15,24..31}
    uint4* dst0 = (uint4*)(&lA[buf][m_l*KB + khalf*8]);        // 16B aligned
    uint4* dst1 = (uint4*)(&lA[buf][m_l*KB + 16 + khalf*8]);   // 16B aligned
    uint4 q0; q0.x = pk[0]; q0.y = pk[1]; q0.z = pk[2]; q0.w = pk[3];
    uint4 q1; q1.x = pk[4]; q1.y = pk[5]; q1.z = pk[6]; q1.w = pk[7];
    *dst0 = q0; *dst1 = q1;
  };

  const v8f zero = {0.f,0.f,0.f,0.f,0.f,0.f,0.f,0.f};
  v8f acc[2][4];
  #pragma unroll
  for (int i = 0; i < 2; ++i)
    #pragma unroll
    for (int j = 0; j < 4; ++j) acc[i][j] = zero;

  // ---- prologue: stage tile 0, wait for params + tile 0 ----
  asyncB(0, 0);
  synthA(0, 0);
  asm volatile("s_wait_asynccnt 0" ::: "memory");
  __syncthreads();

  for (int kt = 0; kt < NK; ++kt) {
    const int cur = kt & 1;
    const int nxt = cur ^ 1;

    // pull current fragments into registers (A: 2x ds_load_b128 each; B: 16 strided 16-bit loads; no shuffles)
    v16h afr[2], bfr[4];
    #pragma unroll
    for (int fm = 0; fm < 2; ++fm) {
      const int row = wm*32 + fm*16 + (lane & 15);
      afr[fm] = *(const v16h*)(&lA[cur][row*KB + ksel]);          // 32B aligned
    }
    #pragma unroll
    for (int fn = 0; fn < 4; ++fn) {
      #pragma unroll
      for (int s = 0; s < 16; ++s)
        bfr[fn][s] = lB[cur][((s & 7) + ((lane >> 4) << 3) + ((s >> 3) << 4))*NT + wn*64 + fn*16 + (lane & 15)];  // n = lane%16, k in table order
    }

    // overlap: stage next tile while the matrix pipe chews on this one
    if (kt + 1 < NK) {
      asyncB(kt + 1, nxt);
      synthA(kt + 1, nxt);   // v_sin VALU work co-executes with WMMA
    }

    #pragma unroll
    for (int fn = 0; fn < 4; ++fn) {
      acc[0][fn] = __builtin_amdgcn_wmma_f32_16x16x32_f16(
          false, afr[0], false, bfr[fn], (short)0, acc[0][fn], false, false);
      acc[1][fn] = __builtin_amdgcn_wmma_f32_16x16x32_f16(
          false, afr[1], false, bfr[fn], (short)0, acc[1][fn], false, false);
    }

    asm volatile("s_wait_asynccnt 0" ::: "memory");
    __syncthreads();
  }

  // ---- write out: v8f register index walks contiguous M (fast dim of (b,k,h,w)) ----
  #pragma unroll
  for (int fm = 0; fm < 2; ++fm) {
    const int m_base = p0 + wm*32 + fm*16 + ((lane >> 4) << 3);
    #pragma unroll
    for (int fn = 0; fn < 4; ++fn) {
      const int n_g = n0 + wn*64 + fn*16 + (lane & 15);
      float* o = out + ((size_t)(b*C_ + n_g) << 12) + m_base;
      v4f lo = {acc[fm][fn][0], acc[fm][fn][1], acc[fm][fn][2], acc[fm][fn][3]};
      v4f hi = {acc[fm][fn][4], acc[fm][fn][5], acc[fm][fn][6], acc[fm][fn][7]};
      *(v4f*)o       = lo;
      *(v4f*)(o + 4) = hi;
    }
  }
}

extern "C" void kernel_launch(void* const* d_in, const int* in_sizes, int n_in,
                              void* d_out, int out_size, void* d_ws, size_t ws_size,
                              hipStream_t stream) {
  const float* w      = (const float*)d_in[0];   // (16,512)
  const float* freqs  = (const float*)d_in[1];   // (1024,2)
  const float* phases = (const float*)d_in[2];   // (1024,)
  const float* weight = (const float*)d_in[3];   // (1024,1024)
  const float* aw     = (const float*)d_in[4];   // (4,512)
  const float* ab     = (const float*)d_in[5];   // (4,)
  float* out = (float*)d_out;                    // (16,1024,64,64)
  float* ws  = (float*)d_ws;
  _Float16* wT = (_Float16*)(ws + OFF_WT);       // 1024x1024 f16, transposed+scaled

  k_affine<<<1, 64, 0, stream>>>(w, aw, ab, ws + OFF_PAR);
  k_chan<<<(B_*C_ + 255)/256, 256, 0, stream>>>(freqs, phases, ws + OFF_PAR, ws);
  k_wt<<<dim3(C_/32, C_/32), 256, 0, stream>>>(weight, wT);

  dim3 grid(C_/NT, NPIX/MT, B_);   // 8 x 32 x 16 = 4096 workgroups
  k_main<<<grid, 256, 0, stream>>>(ws, wT, out);
}
